// MAMBAModel_11347303596382
// MI455X (gfx1250) — matrix operand order repaired and checked by reading, not compiled, not run on hardware
//
#include <hip/hip_runtime.h>
#include <cstdint>
#include <cstddef>

#define DEVINL __device__ __forceinline__

typedef __attribute__((ext_vector_type(16))) __bf16           v16bf;
typedef __attribute__((ext_vector_type(8)))  float            v8f;
typedef __attribute__((ext_vector_type(8)))  unsigned short   u16x8;
typedef __attribute__((ext_vector_type(4)))  unsigned int     u32x4;
typedef __attribute__((ext_vector_type(8)))  int              i32x8;
typedef __attribute__((ext_vector_type(4)))  int              i32x4;

static constexpr int BSZ    = 4;
static constexpr int LSEQ   = 2048;
static constexpr int DIN    = 64;
static constexpr int DMODEL = 1024;
static constexpr int DINNER = 2048;   // EXPAND * D_MODEL
static constexpr int DSTATE = 16;
static constexpr int DTRANK = 64;
static constexpr int DOUTP  = 64;
static constexpr int MROWS  = BSZ * LSEQ;          // 8192 token rows
static constexpr int NDBC   = DTRANK + 2 * DSTATE; // 96

// TDM chunking for the weight (B) operand
static constexpr int KC      = 128;           // K elements per TDM chunk
static constexpr int LDS_PAD = 8;             // 16B pad per 256B row (4 DWORDs)
static constexpr int BROW    = KC + LDS_PAD;  // LDS row pitch in elements (272B)

union Frag16 { u16x8 h[2]; v16bf v; };

DEVINL unsigned short f32_to_bf16(float f) {
  unsigned int u = __builtin_bit_cast(unsigned int, f);
  u += 0x7FFFu + ((u >> 16) & 1u);          // round-to-nearest-even
  return (unsigned short)(u >> 16);
}

// A fragment: 16x32 bf16, M = lane&15.
// lanes 0-15 hold K = 0..15, lanes 16-31 hold K = 16..31.
DEVINL v16bf load_frag_a(const unsigned short* __restrict__ A, int lda,
                         int m0, int k0, int lane) {
  const int row = m0 + (lane & 15);
  const int kh  = (lane & 16) ? 16 : 0;
  const unsigned short* p = A + (size_t)row * lda + (k0 + kh);
  Frag16 f;
  f.h[0] = *(const u16x8*)(p);
  f.h[1] = *(const u16x8*)(p + 8);
  return f.v;
}

// B fragment from global: 32x16 bf16, N = lane&15, lanes hi hold K+16.
DEVINL v16bf load_frag_b_g(const unsigned short* __restrict__ W, int ldw,
                           int n0, int k0, int lane) {
  const int col = n0 + (lane & 15);
  const int kh  = (lane & 16) ? 16 : 0;
  const unsigned short* p = W + (size_t)col * ldw + (k0 + kh);
  Frag16 f;
  f.h[0] = *(const u16x8*)(p);
  f.h[1] = *(const u16x8*)(p + 8);
  return f.v;
}

// B fragment from the TDM-staged LDS tile (row pitch BROW elements).
DEVINL v16bf load_frag_b_lds(const unsigned short* bs, int n_local, int k_local,
                             int lane) {
  const int col = n_local + (lane & 15);
  const int kh  = (lane & 16) ? 16 : 0;
  const unsigned short* p = bs + col * BROW + k_local + kh;
  Frag16 f;
  f.h[0] = *(const u16x8*)(p);      // ds_load_b128
  f.h[1] = *(const u16x8*)(p + 8);
  return f.v;
}

// Issue one TDM 2D tile load: 32 rows x KC cols of bf16 from W-strip into LDS,
// with LDS padding (4 DWORDs per 64 DWORDs) for bank-conflict-free ds reads.
// D# packing per CDNA5 ISA. 6-arg builtin (clang-23 lane).
DEVINL void tdm_load_b_tile(const unsigned short* gsrc, unsigned lds_off,
                            int ldw_elems, int krem_elems) {
  unsigned long long ga = (unsigned long long)(uintptr_t)gsrc;
  u32x4 g0;
  g0[0] = 1u;                                              // count=1, user desc
  g0[1] = lds_off;                                         // lds_addr (bytes)
  g0[2] = (unsigned)ga;                                    // global_addr lo
  g0[3] = (unsigned)((ga >> 32) & 0x01FFFFFFu) | (2u << 30); // addr hi | type=2
  i32x8 g1;
  // data_size=2B (enc 1), pad_enable, pad_interval=64 DWORDs (enc 5),
  // pad_amount=4 DWORDs (enc 3), no multicast / iterate / atomic-barrier.
  g1[0] = (1 << 16) | (1 << 20) | (5 << 22) | (3 << 25);
  const unsigned td0 = (unsigned)krem_elems;               // tensor_dim0
  const unsigned td1 = 32u;                                // tensor_dim1 (rows)
  g1[1] = (int)((td0 & 0xFFFFu) << 16);                    // [15:0] atomic addr=0
  g1[2] = (int)((td0 >> 16) | ((td1 & 0xFFFFu) << 16));
  g1[3] = (int)((td1 >> 16) | ((unsigned)KC << 16));       // tile_dim0 = KC
  g1[4] = 32;                                              // tile_dim1=32, dim2=0
  g1[5] = ldw_elems;                                       // dim0 stride lo
  g1[6] = 0;                                               // stride hi / stride1
  g1[7] = 0;
  const i32x4 z4 = {0, 0, 0, 0};                           // 2D: groups 2/3 unused
  const i32x8 z8 = {0, 0, 0, 0, 0, 0, 0, 0};
  __builtin_amdgcn_tensor_load_to_lds(g0, g1, z4, z4, z8, 0);
}

// C = A(MxK, bf16) * W(NxK, bf16)^T [+ bias], f32 accumulate via WMMA.
// Block: 256 threads = 8 waves, 4(M) x 2(N); wave tile 64x16; BM=256, BN=32.
// OUT_MODE: 0 = f32 row-major, 1 = bf16 row-major, 2 = f32 (B, N, L) transposed.
// USE_TDM: stage the W strip through LDS with tensor_load_to_lds (requires K%128==0).
template<int OUT_MODE, bool USE_TDM>
__global__ __launch_bounds__(256)
void gemm_bf16_wmma(const unsigned short* __restrict__ A,
                    const unsigned short* __restrict__ W,
                    void* __restrict__ Cout,
                    const float* __restrict__ bias,
                    int N, int K) {
  const int lane = threadIdx.x & 31;
  const int wave = threadIdx.x >> 5;
  const int m0 = blockIdx.y * 256 + (wave >> 1) * 64;
  const int n0 = blockIdx.x * 32  + (wave & 1) * 16;

  v8f acc[4] = {};

  if constexpr (USE_TDM) {
    __shared__ unsigned short smemB[2][32 * BROW];
    const int n0_local = (wave & 1) * 16;
    const unsigned short* Wtile = W + (size_t)(blockIdx.x * 32) * K;

    // Prologue: kick off chunk 0 and pre-load first A fragments.
    if (wave == 0)
      tdm_load_b_tile(Wtile, (unsigned)(uintptr_t)&smemB[0][0], K, K);
    v16bf a_cur[4];
#pragma unroll
    for (int i = 0; i < 4; ++i) a_cur[i] = load_frag_a(A, K, m0 + i * 16, 0, lane);

    int buf = 0;
    for (int kc = 0; kc < K; kc += KC) {
      if (wave == 0) __builtin_amdgcn_s_wait_tensorcnt(0);  // chunk kc landed
      __syncthreads();
      if (wave == 0 && kc + KC < K)                          // prefetch next chunk
        tdm_load_b_tile(Wtile + (size_t)(kc + KC),
                        (unsigned)(uintptr_t)&smemB[buf ^ 1][0], K, K - (kc + KC));
      const unsigned short* bs = &smemB[buf][0];
#pragma unroll
      for (int k0 = 0; k0 < KC; k0 += 32) {
        v16bf bf = load_frag_b_lds(bs, n0_local, k0, lane);
        const int knext = kc + k0 + 32;
        v16bf a_nxt[4];
        if (knext < K) {                // software pipeline: fetch A for step k+1
#pragma unroll
          for (int i = 0; i < 4; ++i)
            a_nxt[i] = load_frag_a(A, K, m0 + i * 16, knext, lane);
        }
#pragma unroll
        for (int i = 0; i < 4; ++i)
          acc[i] = __builtin_amdgcn_wmma_f32_16x16x32_bf16(
              false, a_cur[i], false, bf, (short)0, acc[i], false, false);
        if (knext < K) {
#pragma unroll
          for (int i = 0; i < 4; ++i) a_cur[i] = a_nxt[i];
        }
      }
      buf ^= 1;
    }
  } else {
    // Small-K path (K == 64): direct global fragment loads.
    for (int k0 = 0; k0 < K; k0 += 32) {
      v16bf bf = load_frag_b_g(W, K, n0, k0, lane);
#pragma unroll
      for (int i = 0; i < 4; ++i) {
        v16bf af = load_frag_a(A, K, m0 + i * 16, k0, lane);
        acc[i] = __builtin_amdgcn_wmma_f32_16x16x32_bf16(
            false, af, false, bf, (short)0, acc[i], false, false);
      }
    }
  }

  // C layout: VGPR r -> M = r + 8*(lane>=16), N = lane&15
  const int col   = n0 + (lane & 15);
  const int rbase = (lane & 16) ? 8 : 0;
  const float bv  = bias ? bias[col] : 0.0f;
#pragma unroll
  for (int i = 0; i < 4; ++i) {
#pragma unroll
    for (int r = 0; r < 8; ++r) {
      const int row = m0 + i * 16 + rbase + r;
      const float v = acc[i][r] + bv;
      if (OUT_MODE == 0) {
        ((float*)Cout)[(size_t)row * N + col] = v;
      } else if (OUT_MODE == 1) {
        ((unsigned short*)Cout)[(size_t)row * N + col] = f32_to_bf16(v);
      } else {
        const int b = row / LSEQ, l = row % LSEQ;
        ((float*)Cout)[((size_t)b * N + col) * LSEQ + l] = v;
      }
    }
  }
}

// Generic f32 -> bf16 conversion (for weight matrices).
__global__ void cvt_bf16_kernel(const float* __restrict__ in,
                                unsigned short* __restrict__ out, int n) {
  int i = blockIdx.x * blockDim.x + threadIdx.x;
  if (i < n) out[i] = f32_to_bf16(in[i]);
}

// x (B, DIN, L) f32 -> xt (B*L, DIN) bf16 (token-major for GEMM A operand)
__global__ void transpose_x_kernel(const float* __restrict__ x,
                                   unsigned short* __restrict__ xt) {
  int idx = blockIdx.x * blockDim.x + threadIdx.x;   // B*L*DIN
  int i  = idx % DIN;
  int bl = idx / DIN;
  int l  = bl % LSEQ;
  int b  = bl / LSEQ;
  xt[idx] = f32_to_bf16(x[((size_t)b * DIN + i) * LSEQ + l]);
}

// dbc (M, 96) f32 -> dt (M, 64) bf16 (first DTRANK cols)
__global__ void dt_slice_kernel(const float* __restrict__ dbc,
                                unsigned short* __restrict__ dt) {
  int idx = blockIdx.x * blockDim.x + threadIdx.x;   // M * DTRANK
  int r = idx & (DTRANK - 1);
  int m = idx >> 6;
  dt[idx] = f32_to_bf16(dbc[(size_t)m * NDBC + r]);
}

// Depthwise causal conv (k=4) + bias + SiLU over xi = xz[:, :DINNER].
__global__ void conv_silu_kernel(const float* __restrict__ xz,
                                 const float* __restrict__ conv_w,
                                 const float* __restrict__ conv_b,
                                 float* __restrict__ u_f,
                                 unsigned short* __restrict__ u_h) {
  int idx = blockIdx.x * blockDim.x + threadIdx.x;   // M * DINNER
  int d = idx & (DINNER - 1);
  int m = idx >> 11;
  int l = m & (LSEQ - 1);
  float acc = conv_b[d];
#pragma unroll
  for (int j = 0; j < 4; ++j) {
    int ll = l - 3 + j;
    if (ll >= 0)
      acc += conv_w[d * 4 + j] * xz[(size_t)(m - 3 + j) * (2 * DINNER) + d];
  }
  float s = acc / (1.0f + __expf(-acc));             // SiLU
  u_f[idx] = s;
  u_h[idx] = f32_to_bf16(s);
}

// Selective scan: one thread per (b, d). 16-state recurrence in registers.
__global__ __launch_bounds__(256)
void scan_kernel(const float* __restrict__ dpre,
                 const float* __restrict__ dt_b,
                 const float* __restrict__ A_log,
                 const float* __restrict__ Dvec,
                 const float* __restrict__ u_f,
                 const float* __restrict__ xz,
                 const float* __restrict__ dbc,
                 unsigned short* __restrict__ y_h) {
  int t = blockIdx.x * blockDim.x + threadIdx.x;    // B * DINNER
  int d = t & (DINNER - 1);
  int b = t >> 11;

  float Aneg[DSTATE], h[DSTATE];
#pragma unroll
  for (int s = 0; s < DSTATE; ++s) {
    Aneg[s] = -__expf(A_log[(size_t)d * DSTATE + s]);
    h[s] = 0.0f;
  }
  const float Dd  = Dvec[d];
  const float dtb = dt_b[d];

  for (int l = 0; l < LSEQ; ++l) {
    const size_t m = (size_t)b * LSEQ + l;
    float dp = dpre[m * DINNER + d] + dtb;
    float delta = (dp > 20.0f) ? dp : log1pf(__expf(dp));
    float uv = u_f[m * DINNER + d];
    float zv = xz[m * (2 * DINNER) + DINNER + d];
    float du = delta * uv;
    const float* bc = dbc + m * NDBC + DTRANK;      // B at [0,16), C at [16,32)
    float y = 0.0f;
#pragma unroll
    for (int s = 0; s < DSTATE; ++s) {
      float dA = __expf(delta * Aneg[s]);
      h[s] = dA * h[s] + du * bc[s];
      y += h[s] * bc[DSTATE + s];
    }
    y = (y + uv * Dd) * (zv / (1.0f + __expf(-zv)));
    y_h[m * DINNER + d] = f32_to_bf16(y);
  }
}

extern "C" void kernel_launch(void* const* d_in, const int* in_sizes, int n_in,
                              void* d_out, int out_size, void* d_ws, size_t ws_size,
                              hipStream_t stream) {
  const float* x        = (const float*)d_in[0];
  const float* enc_w    = (const float*)d_in[1];
  const float* enc_b    = (const float*)d_in[2];
  const float* in_proj  = (const float*)d_in[3];
  const float* conv_w   = (const float*)d_in[4];
  const float* conv_b   = (const float*)d_in[5];
  const float* x_proj   = (const float*)d_in[6];
  const float* dt_proj  = (const float*)d_in[7];
  const float* dt_b     = (const float*)d_in[8];
  const float* A_log    = (const float*)d_in[9];
  const float* Dvec     = (const float*)d_in[10];
  const float* out_proj = (const float*)d_in[11];
  const float* dec_w    = (const float*)d_in[12];
  const float* dec_b    = (const float*)d_in[13];
  (void)in_sizes; (void)n_in; (void)out_size; (void)ws_size;

  char* ws = (char*)d_ws;
  size_t off = 0;
  auto alloc = [&](size_t bytes) -> char* {
    char* p = ws + off;
    off += (bytes + 255) & ~(size_t)255;
    return p;
  };

  unsigned short* xt     = (unsigned short*)alloc((size_t)MROWS * DIN * 2);
  unsigned short* encw_h = (unsigned short*)alloc((size_t)DMODEL * DIN * 2);
  unsigned short* inpw_h = (unsigned short*)alloc((size_t)2 * DINNER * DMODEL * 2);
  unsigned short* xpw_h  = (unsigned short*)alloc((size_t)NDBC * DINNER * 2);
  unsigned short* dtpw_h = (unsigned short*)alloc((size_t)DINNER * DTRANK * 2);
  unsigned short* outw_h = (unsigned short*)alloc((size_t)DMODEL * DINNER * 2);
  unsigned short* decw_h = (unsigned short*)alloc((size_t)DOUTP * DMODEL * 2);
  unsigned short* h_h    = (unsigned short*)alloc((size_t)MROWS * DMODEL * 2);
  float*          xz     = (float*)         alloc((size_t)MROWS * 2 * DINNER * 4);
  float*          u_f    = (float*)         alloc((size_t)MROWS * DINNER * 4);
  unsigned short* u_h    = (unsigned short*)alloc((size_t)MROWS * DINNER * 2);
  float*          dbc    = (float*)         alloc((size_t)MROWS * NDBC * 4);
  unsigned short* dt_h   = (unsigned short*)alloc((size_t)MROWS * DTRANK * 2);
  float*          dpre   = (float*)         alloc((size_t)MROWS * DINNER * 4);
  unsigned short* y_h    = (unsigned short*)alloc((size_t)MROWS * DINNER * 2);
  unsigned short* out_h  = (unsigned short*)alloc((size_t)MROWS * DMODEL * 2);

  const dim3 blk(256);
  auto cvt = [&](const float* src, unsigned short* dst, int n) {
    cvt_bf16_kernel<<<dim3((n + 255) / 256), blk, 0, stream>>>(src, dst, n);
  };

  // Stage 0: layout/precision conversions
  transpose_x_kernel<<<dim3((MROWS * DIN) / 256), blk, 0, stream>>>(x, xt);
  cvt(enc_w,    encw_h, DMODEL * DIN);
  cvt(in_proj,  inpw_h, 2 * DINNER * DMODEL);
  cvt(x_proj,   xpw_h,  NDBC * DINNER);
  cvt(dt_proj,  dtpw_h, DINNER * DTRANK);
  cvt(out_proj, outw_h, DMODEL * DINNER);
  cvt(dec_w,    decw_h, DOUTP * DMODEL);

  // Stage 1: h = x^T @ enc_w^T + enc_b (K=64, direct path) -> bf16
  gemm_bf16_wmma<1, false><<<dim3(DMODEL / 32, MROWS / 256), blk, 0, stream>>>(
      xt, encw_h, h_h, enc_b, DMODEL, DIN);

  // Stage 2: xz = h @ in_proj^T (K=1024, TDM path) -> f32
  gemm_bf16_wmma<0, true><<<dim3((2 * DINNER) / 32, MROWS / 256), blk, 0, stream>>>(
      h_h, inpw_h, xz, nullptr, 2 * DINNER, DMODEL);

  // Stage 3: depthwise causal conv + SiLU -> u (f32 + bf16)
  conv_silu_kernel<<<dim3((MROWS * DINNER) / 256), blk, 0, stream>>>(
      xz, conv_w, conv_b, u_f, u_h);

  // Stage 4: dbc = u @ x_proj^T (K=2048, TDM path) -> f32 (M, 96)
  gemm_bf16_wmma<0, true><<<dim3(NDBC / 32, MROWS / 256), blk, 0, stream>>>(
      u_h, xpw_h, dbc, nullptr, NDBC, DINNER);

  // Stage 5: dt slice -> bf16; dpre = dt @ dt_proj^T (K=64, direct) -> f32
  dt_slice_kernel<<<dim3((MROWS * DTRANK) / 256), blk, 0, stream>>>(dbc, dt_h);
  gemm_bf16_wmma<0, false><<<dim3(DINNER / 32, MROWS / 256), blk, 0, stream>>>(
      dt_h, dtpw_h, dpre, nullptr, DINNER, DTRANK);

  // Stage 6: selective scan -> y bf16
  scan_kernel<<<dim3((BSZ * DINNER) / 256), blk, 0, stream>>>(
      dpre, dt_b, A_log, Dvec, u_f, xz, dbc, y_h);

  // Stage 7: out = y @ out_proj^T (K=2048, TDM path) -> bf16
  gemm_bf16_wmma<1, true><<<dim3(DMODEL / 32, MROWS / 256), blk, 0, stream>>>(
      y_h, outw_h, out_h, nullptr, DMODEL, DINNER);

  // Stage 8: dec = out @ dec_w^T + dec_b (K=1024, TDM path), stored (B, 64, L)
  gemm_bf16_wmma<2, true><<<dim3(DOUTP / 32, MROWS / 256), blk, 0, stream>>>(
      out_h, decw_h, d_out, dec_b, DOUTP, DMODEL);
}
